// CrossNetMix_50663434224288
// MI455X (gfx1250) — matrix operand order repaired and checked by reading, not compiled, not run on hardware
//
#include <hip/hip_runtime.h>

// ---------------------------------------------------------------------------
// CrossNetMix for MI455X (gfx1250): all four GEMMs (gate, V, C, U^T) on
// v_wmma_f32_16x16x32_bf16, f32 accumulate, weights pre-packed into WMMA
// B-fragment order. B=16384, IN_C=512, RANK=64, E=4, 3 cross layers.
// ---------------------------------------------------------------------------

typedef __attribute__((ext_vector_type(16))) __bf16 v16bf;
typedef __attribute__((ext_vector_type(8)))  float  v8f;

#define B_ROWS 16384
#define IN_C   512
#define RANK   64
#define NEXP   4
#define NCROSS 3

#define XI_STRIDE 520   // 512 + 8 halves: breaks LDS bank conflicts on b128 loads
#define V_STRIDE  72    // 64 + 8 halves

#define NV_PACK (NCROSS*NEXP*IN_C*RANK)   // 393216
#define NC_PACK (NCROSS*NEXP*RANK*RANK)   // 49152
#define NU_PACK (NCROSS*NEXP*IN_C*RANK)   // 393216
#define NG_PACK (16*32*16)                // 8192 (gate_w padded to N=16)

#if __has_builtin(__builtin_amdgcn_tanhf)
#define TANH_F32(x) __builtin_amdgcn_tanhf(x)
#elif __has_builtin(__builtin_amdgcn_tanh_f32)
#define TANH_F32(x) __builtin_amdgcn_tanh_f32(x)
#else
#define TANH_F32(x) tanhf(x)
#endif

static __device__ __forceinline__ unsigned short f2bf(float f) {
  unsigned int u = __builtin_bit_cast(unsigned int, f);
  u += 0x7FFFu + ((u >> 16) & 1u);            // round-to-nearest-even
  return (unsigned short)(u >> 16);
}

// ---------------------------------------------------------------------------
// Pack all weights (f32 -> bf16) into WMMA B-fragment order:
//   fragment order: [...][kstep][ntile][lane(32)][16 halves]
//   lane l holds column n = nt*16 + (l&15); halves j cover
//   k = ks*32 + ((l&16)?16:0) + j   (same contiguous k order as the A loads)
// gate_w [4,512] is packed as one zero-padded 512x16 B panel (layer-invariant).
// ---------------------------------------------------------------------------
__global__ void pack_weights_kernel(const float* __restrict__ Us,
                                    const float* __restrict__ Vs,
                                    const float* __restrict__ Cs,
                                    const float* __restrict__ gate_w,
                                    unsigned short* __restrict__ pV,
                                    unsigned short* __restrict__ pC,
                                    unsigned short* __restrict__ pU,
                                    unsigned short* __restrict__ pG) {
  int tid = blockIdx.x * blockDim.x + threadIdx.x;
  if (tid < NV_PACK) {
    // pV: [L][E][ks:16][nt:4][lane:32][16]
    int p = tid;
    int j = p & 15, lane = (p >> 4) & 31, nt = (p >> 9) & 3, ks = (p >> 11) & 15;
    int e = (p >> 15) & 3, layer = p >> 17;
    int k = ks * 32 + ((lane & 16) ? 16 : 0) + j;     // 0..511
    int n = nt * 16 + (lane & 15);                    // 0..63
    pV[p] = f2bf(Vs[(((long long)layer * NEXP + e) * IN_C + k) * RANK + n]);
  } else if (tid < NV_PACK + NC_PACK) {
    // pC: [L][E][ks:2][nt:4][lane:32][16]
    int p = tid - NV_PACK;
    int j = p & 15, lane = (p >> 4) & 31, nt = (p >> 9) & 3, ks = (p >> 11) & 1;
    int e = (p >> 12) & 3, layer = p >> 14;
    int k = ks * 32 + ((lane & 16) ? 16 : 0) + j;     // 0..63
    int n = nt * 16 + (lane & 15);                    // 0..63
    pC[p] = f2bf(Cs[(((long long)layer * NEXP + e) * RANK + k) * RANK + n]);
  } else if (tid < NV_PACK + NC_PACK + NU_PACK) {
    // pU (U^T): [L][E][ks:2][nt:32][lane:32][16];  B[k][n] = U[n][k]
    int p = tid - NV_PACK - NC_PACK;
    int j = p & 15, lane = (p >> 4) & 31, nt = (p >> 9) & 31, ks = (p >> 14) & 1;
    int e = (p >> 15) & 3, layer = p >> 17;
    int k = ks * 32 + ((lane & 16) ? 16 : 0) + j;     // 0..63
    int n = nt * 16 + (lane & 15);                    // 0..511
    pU[p] = f2bf(Us[(((long long)layer * NEXP + e) * IN_C + n) * RANK + k]);
  } else if (tid < NV_PACK + NC_PACK + NU_PACK + NG_PACK) {
    // pG: [ks:16][lane:32][16];  B[k][n] = (n<4) ? gate_w[n][k] : 0
    int p = tid - NV_PACK - NC_PACK - NU_PACK;
    int j = p & 15, lane = (p >> 4) & 31, ks = p >> 9;
    int k = ks * 32 + ((lane & 16) ? 16 : 0) + j;     // 0..511
    int n = lane & 15;                                // 0..15 (4..15 padded)
    pG[p] = (n < NEXP) ? f2bf(gate_w[(long long)n * IN_C + k]) : (unsigned short)0;
  }
}

// ---------------------------------------------------------------------------
// One cross layer. grid = B/16 workgroups, 256 threads (8 wave32) each.
// Each workgroup owns a 16-row batch tile.
// ---------------------------------------------------------------------------
__global__ __launch_bounds__(256)
void cross_layer_kernel(const float* __restrict__ x0,
                        const float* __restrict__ xin,
                        float* __restrict__ xout,
                        const unsigned short* __restrict__ pV,
                        const unsigned short* __restrict__ pC,
                        const unsigned short* __restrict__ pU,
                        const unsigned short* __restrict__ pG,
                        const float* __restrict__ biases,
                        int layer) {
  __shared__ __attribute__((aligned(16))) unsigned short sXi[16 * XI_STRIDE];
  __shared__ __attribute__((aligned(16))) unsigned short sV [NEXP * 16 * V_STRIDE];
  __shared__ __attribute__((aligned(16))) unsigned short sCV[NEXP * 16 * V_STRIDE];
  __shared__ float sGate[16 * NEXP];

  const int tid  = threadIdx.x;
  const int lane = tid & 31;
  const int w    = tid >> 5;                  // wave id 0..7
  const int tileRow = blockIdx.x * 16;
  const float* xinTile = xin + (long long)tileRow * IN_C;

  // A-fragment lane geometry (16-bit A, 16x32):
  const int mA  = lane & 15;
  const int hiA = (lane & 16) ? 16 : 0;
  // D/C lane geometry (f32 16x16): VGPR r -> row r + m0D, col nB
  const int nB  = lane & 15;
  const int m0D = (lane & 16) ? 8 : 0;

  // ---- Stage 0: stage xi tile (bf16) into LDS ----
  for (int idx = tid; idx < 16 * IN_C; idx += 256) {
    int m = idx >> 9, c = idx & 511;
    sXi[m * XI_STRIDE + c] = f2bf(xinTile[m * IN_C + c]);
  }
  __syncthreads();

  // ---- Stage 0b (wave 0): gate = xi @ gate_w^T via WMMA (N padded 4->16) ----
  if (w == 0) {                                // wave-uniform branch: EXEC all 1s
    v8f g = {};
#pragma unroll
    for (int ks = 0; ks < 16; ++ks) {
      union { v16bf v; uint4 q[2]; } a, b;
      const unsigned short* ar = &sXi[mA * XI_STRIDE + ks * 32];
      a.q[0] = *(const uint4*)(ar + hiA);
      a.q[1] = *(const uint4*)(ar + 8 + hiA);
      b.v = *(const v16bf*)(pG + (size_t)ks * 512 + (size_t)lane * 16);
      g = __builtin_amdgcn_wmma_f32_16x16x32_bf16(false, a.v, false, b.v,
                                                  (short)0, g, false, false);
    }
    if (nB < NEXP) {                           // lanes holding real experts
#pragma unroll
      for (int r = 0; r < 8; ++r) sGate[(r + m0D) * NEXP + nB] = g[r];
    }
  }

  // ---- Stage 1: v = tanh(xi @ V[e])   K=512 (16 WMMA), 16 tasks / 8 waves ----
#pragma unroll
  for (int t = 0; t < 2; ++t) {
    const int task = w + t * 8;                // uniform per wave, no bound check
    const int e = task >> 2, nt = task & 3;
    const unsigned short* bp =
        pV + ((size_t)(layer * NEXP + e)) * 32768 + (size_t)nt * 512 + (size_t)lane * 16;
    __builtin_prefetch(bp, 0, 1);              // global_prefetch_b8
    v8f acc = {};
#pragma unroll
    for (int ks = 0; ks < 16; ++ks) {
      union { v16bf v; uint4 q[2]; } a, b;
      const unsigned short* ar = &sXi[mA * XI_STRIDE + ks * 32];
      a.q[0] = *(const uint4*)(ar + hiA);
      a.q[1] = *(const uint4*)(ar + 8 + hiA);
      b.v = *(const v16bf*)(bp + (size_t)ks * 2048);
      acc = __builtin_amdgcn_wmma_f32_16x16x32_bf16(false, a.v, false, b.v,
                                                    (short)0, acc, false, false);
    }
#pragma unroll
    for (int r = 0; r < 8; ++r)
      sV[(e * 16 + r + m0D) * V_STRIDE + nt * 16 + nB] = f2bf(TANH_F32(acc[r]));
  }
  __syncthreads();

  // ---- Stage 2: cv = tanh(v @ C[e])   K=64 (2 WMMA), 16 tasks / 8 waves ----
#pragma unroll
  for (int t = 0; t < 2; ++t) {
    const int task = w + t * 8;
    const int e = task >> 2, nt = task & 3;
    const unsigned short* bp =
        pC + ((size_t)(layer * NEXP + e)) * 4096 + (size_t)nt * 512 + (size_t)lane * 16;
    v8f acc = {};
#pragma unroll
    for (int ks = 0; ks < 2; ++ks) {
      union { v16bf v; uint4 q[2]; } a, b;
      const unsigned short* ar = &sV[(e * 16 + mA) * V_STRIDE + ks * 32];
      a.q[0] = *(const uint4*)(ar + hiA);
      a.q[1] = *(const uint4*)(ar + 8 + hiA);
      b.v = *(const v16bf*)(bp + (size_t)ks * 2048);
      acc = __builtin_amdgcn_wmma_f32_16x16x32_bf16(false, a.v, false, b.v,
                                                    (short)0, acc, false, false);
    }
#pragma unroll
    for (int r = 0; r < 8; ++r)
      sCV[(e * 16 + r + m0D) * V_STRIDE + nt * 16 + nB] = f2bf(TANH_F32(acc[r]));
  }
  __syncthreads();

  // ---- Stage 3: ucv = cv @ U^T; fused gate/bias/x0/residual combine ----
  // out = x0 * sum_e gate_e*(ucv_e + bias) + xin
  const float* biasL = biases + (long long)layer * IN_C;
#pragma unroll
  for (int t = 0; t < 4; ++t) {
    const int nt = w + t * 8;                  // 32 ntiles over 8 waves
    const int col = nt * 16 + nB;
    const float bn = biasL[col];
    float tot[8] = {0.f};
#pragma unroll
    for (int e = 0; e < NEXP; ++e) {
      const unsigned short* bp =
          pU + ((size_t)(layer * NEXP + e)) * 32768 + (size_t)nt * 512 + (size_t)lane * 16;
      __builtin_prefetch(bp, 0, 1);
      v8f d = {};
#pragma unroll
      for (int ks = 0; ks < 2; ++ks) {
        union { v16bf v; uint4 q[2]; } a, b;
        const unsigned short* ar = &sCV[(e * 16 + mA) * V_STRIDE + ks * 32];
        a.q[0] = *(const uint4*)(ar + hiA);
        a.q[1] = *(const uint4*)(ar + 8 + hiA);
        b.v = *(const v16bf*)(bp + (size_t)ks * 16384);
        d = __builtin_amdgcn_wmma_f32_16x16x32_bf16(false, a.v, false, b.v,
                                                    (short)0, d, false, false);
      }
#pragma unroll
      for (int r = 0; r < 8; ++r)
        tot[r] += sGate[(r + m0D) * NEXP + e] * (d[r] + bn);
    }
#pragma unroll
    for (int r = 0; r < 8; ++r) {
      const long long idx = (long long)(tileRow + r + m0D) * IN_C + col;
      xout[idx] = x0[idx] * tot[r] + xin[idx];
    }
  }
}

// ---------------------------------------------------------------------------
// Launch: pack weights once per call, then 3 sequential cross layers.
// Ping-pong: L0 x -> d_out, L1 d_out -> ws, L2 ws -> d_out.
// ws layout (bytes): pV @0 (786,432) | pC @1,572,864 (98,304) |
//   pU @1,671,168 (786,432) | pG @3,244,032 (16,384) | xi_tmp @3,260,416
// ---------------------------------------------------------------------------
extern "C" void kernel_launch(void* const* d_in, const int* in_sizes, int n_in,
                              void* d_out, int out_size, void* d_ws, size_t ws_size,
                              hipStream_t stream) {
  (void)in_sizes; (void)n_in; (void)out_size; (void)ws_size;
  const float* x      = (const float*)d_in[0];
  const float* Us     = (const float*)d_in[1];
  const float* Vs     = (const float*)d_in[2];
  const float* Cs     = (const float*)d_in[3];
  const float* gate_w = (const float*)d_in[4];
  const float* biases = (const float*)d_in[5];
  float* out = (float*)d_out;

  char* ws = (char*)d_ws;
  unsigned short* pV = (unsigned short*)(ws);
  unsigned short* pC = (unsigned short*)(ws + 1572864);
  unsigned short* pU = (unsigned short*)(ws + 1671168);
  unsigned short* pG = (unsigned short*)(ws + 3244032);
  float*         tmp = (float*)(ws + 3260416);

  const int totalPack = NV_PACK + NC_PACK + NU_PACK + NG_PACK;
  pack_weights_kernel<<<(totalPack + 255) / 256, 256, 0, stream>>>(
      Us, Vs, Cs, gate_w, pV, pC, pU, pG);

  dim3 grid(B_ROWS / 16);
  cross_layer_kernel<<<grid, 256, 0, stream>>>(x, x,   out, pV, pC, pU, pG, biases, 0);
  cross_layer_kernel<<<grid, 256, 0, stream>>>(x, out, tmp, pV, pC, pU, pG, biases, 1);
  cross_layer_kernel<<<grid, 256, 0, stream>>>(x, tmp, out, pV, pC, pU, pG, biases, 2);
}
